// TemporalRelationGraph_48344151884421
// MI455X (gfx1250) — matrix operand order repaired and checked by reading, not compiled, not run on hardware
//
#include <hip/hip_runtime.h>

#define NNODES 20000
#define NEDGES 200000
#define ENTOT  220000      // NEDGES + NNODES self loops
#define HEADS  8
#define CH     256
#define HC     2048        // HEADS*CH
#define INCH   256
#define OUTCH  250
#define OUTPAD 256

typedef __attribute__((ext_vector_type(16))) __bf16 v16bf;
typedef __attribute__((ext_vector_type(8)))  __bf16 v8bf;
typedef __attribute__((ext_vector_type(8)))  float  v8f;

// ---------------- helpers ----------------

__device__ __forceinline__ void atomicMaxF(float* addr, float v) {
  // IEEE trick: non-negative floats order as signed ints, negative as inverted unsigned.
  if (v >= 0.0f) atomicMax((int*)addr, __float_as_int(v));
  else           atomicMin((unsigned int*)addr, __float_as_uint(v));
}

// ---------------- init / convert ----------------

__global__ void k_zero_f32(float* __restrict__ p, int n) {
  int i = blockIdx.x * blockDim.x + threadIdx.x;
  int stride = gridDim.x * blockDim.x;
  for (; i < n; i += stride) p[i] = 0.0f;
}

__global__ void k_init_ms(float* __restrict__ m, float* __restrict__ s,
                          float* __restrict__ pooled, int n) {
  int i = blockIdx.x * blockDim.x + threadIdx.x;
  if (i < n) { m[i] = -INFINITY; s[i] = 0.0f; }
  if (i < HEADS) pooled[i] = 0.0f;
}

__global__ void k_f32_to_bf16(const float* __restrict__ in, __bf16* __restrict__ out, int n) {
  int i = blockIdx.x * blockDim.x + threadIdx.x;
  if (i < n) out[i] = (__bf16)in[i];
}

// Pack row-major fp32 W[K=256][Nsrc] into bf16 WMMA B fragments:
// P[((kt*ntiles + nt)*32 + lane)*16 + j] = W[kt*32 + klo(lane) + j + (j>=8 ? 8 : 0)][nt*16 + lane%16]
// klo = 0 for lanes 0-15, 8 for lanes 16-31 (ISA 16-bit B layout, 32x16 tile).
__global__ void k_pack_b(const float* __restrict__ W, __bf16* __restrict__ P,
                         int Nsrc, int Npad) {
  int idx = blockIdx.x * blockDim.x + threadIdx.x;
  int ntiles = Npad >> 4;
  int total = 8 * ntiles * 32;               // K = 256 -> 8 k-tiles of 32
  if (idx >= total) return;
  int lane = idx & 31;
  int t = idx >> 5;
  int nt = t % ntiles;
  int kt = t / ntiles;
  int n = nt * 16 + (lane & 15);
  int kbase = kt * 32 + ((lane & 16) ? 8 : 0);
  __bf16* o = P + (size_t)idx * 16;
#pragma unroll
  for (int j = 0; j < 16; ++j) {
    float v = (n < Nsrc) ? W[(size_t)(kbase + j + ((j >= 8) ? 8 : 0)) * Nsrc + n] : 0.0f;
    o[j] = (__bf16)v;
  }
}

// ---------------- WMMA GEMM (K fixed at 256) ----------------
// A: row-major bf16 [M][256]; Bp: packed fragments; C: fp32 [M][Nstore] (+bias).
// Block = 256 threads = 8 waves; wave -> one 16-row M tile; blockIdx.y -> 4 N tiles (16x64 strip).
__global__ void k_wmma_gemm(const __bf16* __restrict__ A, const __bf16* __restrict__ Bp,
                            float* __restrict__ C, const float* __restrict__ bias,
                            int M, int ntiles, int Nstore) {
  int wave = threadIdx.x >> 5;
  int lane = threadIdx.x & 31;
  int mtile = blockIdx.x * 8 + wave;
  if (mtile * 16 >= M) return;
  int lm = lane & 15;
  int khalf = (lane & 16) ? 8 : 0;           // A: K-half select; D: row-half select
  const __bf16* Arow = A + (size_t)(mtile * 16 + lm) * 256;
  int nt0 = blockIdx.y << 2;
  const __bf16* Bbase = Bp + (size_t)nt0 * 512 + (size_t)lane * 16;

  v8f acc[4] = {};
#pragma unroll
  for (int kt = 0; kt < 8; ++kt) {
    const __bf16* ap = Arow + kt * 32 + khalf;
    v8bf a0 = *(const v8bf*)ap;              // K = base .. base+7
    v8bf a1 = *(const v8bf*)(ap + 16);       // K = base+16 .. base+23
    v16bf av = __builtin_shufflevector(a0, a1,
        0, 1, 2, 3, 4, 5, 6, 7, 8, 9, 10, 11, 12, 13, 14, 15);
    const __bf16* bp = Bbase + (size_t)kt * ntiles * 512;
#pragma unroll
    for (int t = 0; t < 4; ++t) {
      v16bf bv = *(const v16bf*)(bp + t * 512);
      acc[t] = __builtin_amdgcn_wmma_f32_16x16x32_bf16(
          false, av, false, bv, (short)0, acc[t], false, false);
    }
  }

  int rbase = mtile * 16 + khalf;            // D: lanes>=16 hold rows M=8..15
#pragma unroll
  for (int t = 0; t < 4; ++t) {
    int ncol = (nt0 + t) * 16 + lm;
    if (ncol >= Nstore) continue;
    float bv = bias ? bias[ncol] : 0.0f;
#pragma unroll
    for (int r = 0; r < 8; ++r)
      C[(size_t)(rbase + r) * Nstore + ncol] = acc[t][r] + bv;
  }
}

// ---------------- attention scores: a_s[n,h] = dot(xh[n,h,:], att_src[h,:]) ----------------

__global__ void k_attn_scores(const float* __restrict__ xh, const float* __restrict__ att_s,
                              const float* __restrict__ att_d, float* __restrict__ a_s,
                              float* __restrict__ a_d) {
  int n = blockIdx.x;
  int h = threadIdx.x >> 5;
  int lane = threadIdx.x & 31;
  const float* xr = xh + (size_t)n * HC + h * CH;
  float ss = 0.0f, sd = 0.0f;
#pragma unroll
  for (int c = lane; c < CH; c += 32) {
    float v = xr[c];
    ss += v * att_s[h * CH + c];
    sd += v * att_d[h * CH + c];
  }
#pragma unroll
  for (int off = 16; off > 0; off >>= 1) {
    ss += __shfl_down(ss, off, 32);
    sd += __shfl_down(sd, off, 32);
  }
  if (lane == 0) { a_s[n * HEADS + h] = ss; a_d[n * HEADS + h] = sd; }
}

// ---------------- edge passes ----------------

__global__ void k_edge_logits(const int* __restrict__ ei, const float* __restrict__ a_s,
                              const float* __restrict__ a_d, float* __restrict__ ebuf,
                              float* __restrict__ mbuf) {
  int idx = blockIdx.x * blockDim.x + threadIdx.x;
  if (idx >= ENTOT * HEADS) return;
  int e = idx >> 3, h = idx & 7;
  int src = (e < NEDGES) ? ei[e] : (e - NEDGES);
  int dst = (e < NEDGES) ? ei[NEDGES + e] : (e - NEDGES);
  float v = a_s[src * HEADS + h] + a_d[dst * HEADS + h];
  v = (v > 0.0f) ? v : 0.2f * v;             // leaky_relu 0.2
  ebuf[idx] = v;
  atomicMaxF(&mbuf[dst * HEADS + h], v);
}

__global__ void k_edge_exp(const int* __restrict__ ei, const float* __restrict__ mbuf,
                           float* __restrict__ ebuf, float* __restrict__ sbuf) {
  int idx = blockIdx.x * blockDim.x + threadIdx.x;
  if (idx >= ENTOT * HEADS) return;
  int e = idx >> 3, h = idx & 7;
  int dst = (e < NEDGES) ? ei[NEDGES + e] : (e - NEDGES);
  float ex = __expf(ebuf[idx] - mbuf[dst * HEADS + h]);
  ebuf[idx] = ex;
  unsafeAtomicAdd(&sbuf[dst * HEADS + h], ex);
}

// One block per edge; wave h handles head h; 8 channels per thread (float4 x2).
__global__ void k_aggregate(const int* __restrict__ ei, const float* __restrict__ xh,
                            const float* __restrict__ ebuf, const float* __restrict__ sbuf,
                            float* __restrict__ agg) {
  int e = blockIdx.x;
  int src = (e < NEDGES) ? ei[e] : (e - NEDGES);
  int dst = (e < NEDGES) ? ei[NEDGES + e] : (e - NEDGES);
  __shared__ float alpha[HEADS];
  int t = threadIdx.x;
  if (t < HEADS) alpha[t] = ebuf[(size_t)e * HEADS + t] / (sbuf[dst * HEADS + t] + 1e-16f);
  __syncthreads();
  int h = t >> 5;
  int c0 = (t & 31) * 8;
  const float4* xs = (const float4*)(xh + (size_t)src * HC + h * CH + c0);
  float4 v0 = xs[0], v1 = xs[1];
  float al = alpha[h];
  float* ad = agg + (size_t)dst * HC + h * CH + c0;
  unsafeAtomicAdd(ad + 0, v0.x * al);
  unsafeAtomicAdd(ad + 1, v0.y * al);
  unsafeAtomicAdd(ad + 2, v0.z * al);
  unsafeAtomicAdd(ad + 3, v0.w * al);
  unsafeAtomicAdd(ad + 4, v1.x * al);
  unsafeAtomicAdd(ad + 5, v1.y * al);
  unsafeAtomicAdd(ad + 6, v1.z * al);
  unsafeAtomicAdd(ad + 7, v1.w * al);
}

// ---------------- pooling / head softmax / fuse ----------------

__global__ void k_pool(const float* __restrict__ agg, const float* __restrict__ b_gat,
                       float* __restrict__ pooled) {
  int n = blockIdx.x;
  int t = threadIdx.x;
  int h = t >> 5, lane = t & 31;
  int c0 = lane * 8;
  const float4* p = (const float4*)(agg + (size_t)n * HC + h * CH + c0);
  const float4* b = (const float4*)(b_gat + h * CH + c0);
  float4 v0 = p[0], v1 = p[1], g0 = b[0], g1 = b[1];
  float s = (v0.x + g0.x) + (v0.y + g0.y) + (v0.z + g0.z) + (v0.w + g0.w)
          + (v1.x + g1.x) + (v1.y + g1.y) + (v1.z + g1.z) + (v1.w + g1.w);
#pragma unroll
  for (int off = 16; off > 0; off >>= 1) s += __shfl_down(s, off, 32);
  if (lane == 0) unsafeAtomicAdd(&pooled[h], s);
}

__global__ void k_head_weights(const float* __restrict__ pooled, const float* __restrict__ cw,
                               const float* __restrict__ cb, float* __restrict__ wout) {
  if (threadIdx.x != 0 || blockIdx.x != 0) return;
  float g[HEADS];
  float mx = -1e30f;
  const float inv = 1.0f / ((float)NNODES * (float)CH);
  for (int h = 0; h < HEADS; ++h) {
    float p = pooled[h] * inv;
    float gg = p * cw[0] + cb[0];
    gg = gg > 0.0f ? gg : 0.0f;              // relu
    g[h] = gg;
    if (gg > mx) mx = gg;
  }
  float sum = 0.0f;
  for (int h = 0; h < HEADS; ++h) { g[h] = __expf(g[h] - mx); sum += g[h]; }
  for (int h = 0; h < HEADS; ++h) wout[h] = g[h] / sum;
}

__global__ void k_fuse(const float* __restrict__ agg, const float* __restrict__ b_gat,
                       const float* __restrict__ wout, const float* __restrict__ x,
                       __bf16* __restrict__ Hf) {
  int idx = blockIdx.x * blockDim.x + threadIdx.x;
  if (idx >= NNODES * INCH) return;
  int n = idx >> 8, c = idx & 255;
  float f = 0.0f;
#pragma unroll
  for (int h = 0; h < HEADS; ++h)
    f += (agg[(size_t)n * HC + h * CH + c] + b_gat[h * CH + c]) * wout[h];
  f += x[idx];
  f = f > 0.0f ? f : 0.0f;
  Hf[idx] = (__bf16)f;
}

// ---------------- launch ----------------

extern "C" void kernel_launch(void* const* d_in, const int* in_sizes, int n_in,
                              void* d_out, int out_size, void* d_ws, size_t ws_size,
                              hipStream_t stream) {
  const float* x      = (const float*)d_in[0];
  const int*   ei     = (const int*)d_in[1];
  const float* W_gat  = (const float*)d_in[2];
  const float* att_s  = (const float*)d_in[3];
  const float* att_d  = (const float*)d_in[4];
  const float* b_gat  = (const float*)d_in[5];
  const float* conv_w = (const float*)d_in[6];
  const float* conv_b = (const float*)d_in[7];
  const float* W_lin  = (const float*)d_in[8];
  const float* b_lin  = (const float*)d_in[9];
  float* out = (float*)d_out;

  size_t off = 0;
  auto carve = [&](size_t bytes) -> void* {
    void* p = (char*)d_ws + off;
    off += (bytes + 255) & ~(size_t)255;
    return p;
  };
  float*  xh     = (float*) carve((size_t)NNODES * HC * 4);        // 164 MB
  float*  agg    = (float*) carve((size_t)NNODES * HC * 4);        // 164 MB
  __bf16* xbf    = (__bf16*)carve((size_t)NNODES * INCH * 2);      // 10 MB
  __bf16* hbf    = (__bf16*)carve((size_t)NNODES * INCH * 2);      // 10 MB
  __bf16* bp1    = (__bf16*)carve((size_t)8 * 128 * 32 * 16 * 2);  // 1 MB
  __bf16* bp2    = (__bf16*)carve((size_t)8 * 16 * 32 * 16 * 2);
  float*  a_s    = (float*) carve((size_t)NNODES * HEADS * 4);
  float*  a_d    = (float*) carve((size_t)NNODES * HEADS * 4);
  float*  ebuf   = (float*) carve((size_t)ENTOT * HEADS * 4);      // 7 MB
  float*  mbuf   = (float*) carve((size_t)NNODES * HEADS * 4);
  float*  sbuf   = (float*) carve((size_t)NNODES * HEADS * 4);
  float*  pooled = (float*) carve(64);
  float*  wgt    = (float*) carve(64);

  // precision path: bf16 inputs, fp32 WMMA accumulation
  k_f32_to_bf16<<<(NNODES * INCH + 255) / 256, 256, 0, stream>>>(x, xbf, NNODES * INCH);
  k_pack_b<<<(8 * 128 * 32 + 255) / 256, 256, 0, stream>>>(W_gat, bp1, HC, HC);
  k_pack_b<<<(8 * 16 * 32 + 255) / 256, 256, 0, stream>>>(W_lin, bp2, OUTCH, OUTPAD);

  // GEMM1: xh[20000,2048] = x @ W_gat    (M tiles: 1250 -> 157 blocks of 8 waves)
  k_wmma_gemm<<<dim3(157, 32), 256, 0, stream>>>(xbf, bp1, xh, nullptr, NNODES, 128, HC);

  k_attn_scores<<<NNODES, 256, 0, stream>>>(xh, att_s, att_d, a_s, a_d);

  k_zero_f32<<<8192, 256, 0, stream>>>(agg, NNODES * HC);
  k_init_ms<<<(NNODES * HEADS + 255) / 256, 256, 0, stream>>>(mbuf, sbuf, pooled, NNODES * HEADS);

  k_edge_logits<<<(ENTOT * HEADS + 255) / 256, 256, 0, stream>>>(ei, a_s, a_d, ebuf, mbuf);
  k_edge_exp<<<(ENTOT * HEADS + 255) / 256, 256, 0, stream>>>(ei, mbuf, ebuf, sbuf);
  k_aggregate<<<ENTOT, 256, 0, stream>>>(ei, xh, ebuf, sbuf, agg);

  k_pool<<<NNODES, 256, 0, stream>>>(agg, b_gat, pooled);
  k_head_weights<<<1, 32, 0, stream>>>(pooled, conv_w, conv_b, wgt);
  k_fuse<<<(NNODES * INCH + 255) / 256, 256, 0, stream>>>(agg, b_gat, wgt, x, hbf);

  // GEMM2: out[20000,250] = Hf @ W_lin + b_lin (N padded to 256, stores guarded)
  k_wmma_gemm<<<dim3(157, 4), 256, 0, stream>>>(hbf, bp2, out, b_lin, NNODES, 16, OUTCH);
}
